// RotateDeformConv2D_77489799954869
// MI455X (gfx1250) — matrix operand order repaired and checked by reading, not compiled, not run on hardware
//
#include <hip/hip_runtime.h>

typedef __bf16 v16bf __attribute__((ext_vector_type(16)));
typedef float  v8f   __attribute__((ext_vector_type(8)));

#define IMG_H 256
#define IMG_W 256
#define NPIX  65536      // 256*256
#define CIN   64
#define COUT  64

// ---------------------------------------------------------------------------
// Kernel 0: 2x2 average-pool the 4x512x512 group mask down to 4x256x256
// ---------------------------------------------------------------------------
__global__ __launch_bounds__(256) void pool_gm_kernel(const float* __restrict__ G,
                                                      float* __restrict__ gm) {
    int idx = blockIdx.x * 256 + threadIdx.x;     // 4*65536 total
    if (idx >= 4 * NPIX) return;
    int r = idx >> 16;
    int y = (idx >> 8) & 255;
    int x = idx & 255;
    const float* g = G + (size_t)r * 512 * 512;
    int o = (2 * y) * 512 + 2 * x;
    gm[idx] = 0.25f * (g[o] + g[o + 1] + g[o + 512] + g[o + 513]);
}

// ---------------------------------------------------------------------------
// Kernel 1: build rotated filters tw[o*4+r, c, tap], cast to bf16 and store
// pre-swizzled in the CDNA5 16-bit A-fragment layout (ISA 7.12.2):
//   lane l: M = l%16, hi = l/16; elem e -> VGPR v=e/2, half=e%2;
//   K(v<4) = 8*hi + 2v + half ; K(v>=4) = 16 + 8*hi + 2(v-4) + half
// Aw index = (((mt*9 + t)*2 + khalf)*32 + lane)*16 + e
// ---------------------------------------------------------------------------
__global__ __launch_bounds__(256) void build_tw_kernel(const float* __restrict__ weight,
                                                       __bf16* __restrict__ Aw) {
    int idx = blockIdx.x * 256 + threadIdx.x;     // 16*9*2*32*16 = 147456
    if (idx >= 16 * 9 * 2 * 32 * 16) return;
    int e    = idx & 15;
    int r1   = idx >> 4;
    int lane = r1 & 31;
    int r2   = r1 >> 5;
    int h    = r2 & 1;
    int r3   = r2 >> 1;
    int t    = r3 % 9;
    int mt   = r3 / 9;

    int hi = lane >> 4;
    int v  = e >> 1;
    int lo = e & 1;
    int kb = (v < 4) ? (8 * hi + 2 * v + lo) : (16 + 8 * hi + 2 * (v - 4) + lo);
    int c  = 32 * h + kb;                         // K index = input channel
    int M  = mt * 16 + (lane & 15);               // row in [0,256)
    int o  = M >> 2;                              // output channel
    int r  = M & 3;                               // rotation
    int i  = t / 3, j = t % 3;
    int si, sj;
    switch (r) {                                  // np.rot90^r of the 3x3 tap grid
        case 0:  si = i;     sj = j;     break;
        case 1:  si = j;     sj = 2 - i; break;
        case 2:  si = 2 - i; sj = 2 - j; break;
        default: si = 2 - j; sj = i;     break;
    }
    Aw[idx] = (__bf16)weight[((o * CIN + c) * 9) + si * 3 + sj];
}

// ---------------------------------------------------------------------------
// Kernel 2: 3x3 conv producing 18 offset channels + 9 sigmoid(mask) channels.
// One thread per (b,y,x); weight loads are wave-uniform -> SMEM path.
// ---------------------------------------------------------------------------
__global__ __launch_bounds__(256) void conv_off_mask_kernel(
    const float* __restrict__ x, const float* __restrict__ off_w,
    const float* __restrict__ off_b, const float* __restrict__ msk_w,
    const float* __restrict__ msk_b, float* __restrict__ offs,
    float* __restrict__ dm) {
    int gid = blockIdx.x * 256 + threadIdx.x;     // 2*65536
    if (gid >= 2 * NPIX) return;
    int b   = gid >> 16;
    int pix = gid & (NPIX - 1);
    int y   = pix >> 8;
    int xx  = pix & 255;

    float acc[18], accm[9];
#pragma unroll
    for (int oc = 0; oc < 18; ++oc) acc[oc] = off_b[oc];
#pragma unroll
    for (int oc = 0; oc < 9; ++oc) accm[oc] = msk_b[oc];

    const float* xb = x + ((size_t)b << 22);      // b * 64 * 65536
    for (int c = 0; c < CIN; ++c) {
        const float* xc = xb + (c << 16);
        float xv[9];
#pragma unroll
        for (int t = 0; t < 9; ++t) {
            int yy  = y + t / 3 - 1;
            int xx2 = xx + t % 3 - 1;
            bool ok = (yy >= 0) & (yy < IMG_H) & (xx2 >= 0) & (xx2 < IMG_W);
            xv[t] = ok ? xc[(yy << 8) + xx2] : 0.0f;
        }
#pragma unroll
        for (int oc = 0; oc < 18; ++oc) {
            const float* w = off_w + (oc * CIN + c) * 9;
#pragma unroll
            for (int t = 0; t < 9; ++t) acc[oc] = fmaf(w[t], xv[t], acc[oc]);
        }
#pragma unroll
        for (int oc = 0; oc < 9; ++oc) {
            const float* w = msk_w + (oc * CIN + c) * 9;
#pragma unroll
            for (int t = 0; t < 9; ++t) accm[oc] = fmaf(w[t], xv[t], accm[oc]);
        }
    }
#pragma unroll
    for (int oc = 0; oc < 18; ++oc)
        offs[((size_t)(b * 18 + oc) << 16) + pix] = acc[oc];
#pragma unroll
    for (int oc = 0; oc < 9; ++oc)
        dm[((size_t)(b * 9 + oc) << 16) + pix] = 1.0f / (1.0f + __expf(-accm[oc]));
}

// ---------------------------------------------------------------------------
// Kernel 3 (main): one block = 16 waves = 512 threads, 32 pixels per block
// (2 N-subtiles of 16). Wave w owns M-tile w (4 output channels x 4 rots).
// Per tap: two A-fragment loads (one per K-half) feed 4 v_wmma_f32_16x16x32_bf16 (2 K-halves
// x 2 N-subtiles), i.e. 36 WMMAs per wave per block.
// ---------------------------------------------------------------------------
__global__ __launch_bounds__(512) void rotdeform_main_kernel(
    const float* __restrict__ x, const float* __restrict__ offs,
    const float* __restrict__ dm, const __bf16* __restrict__ Aw,
    const float* __restrict__ gm, float* __restrict__ out) {
    // B fragments: [tap][subtile][khalf][lane][elem]; 9*2*2*32*16 bf16 = 36 KB
    __shared__ __attribute__((aligned(32))) __bf16 sB[9 * 2 * 2 * 32 * 16];
    __shared__ int   sPi[9 * 32 * 4];   // 4 corner offsets per (tap, pixel)
    __shared__ float sPw[9 * 32 * 4];   // 4 corner weights (mask folded in)

    const int tid   = threadIdx.x;
    const int pt    = blockIdx.x;       // 4096 blocks of 32 pixels
    const int pbase = pt << 5;
    const int b     = pbase >> 16;
    const int pix   = pbase & (NPIX - 1);
    const int py0   = pix >> 8;
    const int px0   = pix & 255;

    // ---- Phase A: per-(tap, pixel) bilinear parameters -> LDS ----
    if (tid < 288) {
        const int t = tid >> 5;             // tap 0..8
        const int n = tid & 31;             // pixel 0..31
        const int i = t / 3, j = t % 3;
        const int p = pix + n;
        const float off0 = offs[((size_t)(b * 18 + 2 * t) << 16) + p];
        const float off1 = offs[((size_t)(b * 18 + 2 * t + 1) << 16) + p];
        const float sig  = dm[((size_t)(b * 9 + t) << 16) + p];
        const float fy = (float)(py0 + i - 1) + off0;
        const float fx = (float)(px0 + n + j - 1) + off1;
        const float y0f = floorf(fy), x0f = floorf(fx);
        const float ly = fy - y0f, lx = fx - x0f;
        const int y0 = (int)y0f, x0 = (int)x0f;
        const int y1 = y0 + 1,   x1 = x0 + 1;
        const bool vy0 = (y0 >= 0) & (y0 < IMG_H);
        const bool vy1 = (y1 >= 0) & (y1 < IMG_H);
        const bool vx0 = (x0 >= 0) & (x0 < IMG_W);
        const bool vx1 = (x1 >= 0) & (x1 < IMG_W);
        float w00 = (1.0f - ly) * (1.0f - lx) * sig; if (!(vy0 & vx0)) w00 = 0.0f;
        float w01 = (1.0f - ly) * lx          * sig; if (!(vy0 & vx1)) w01 = 0.0f;
        float w10 = ly * (1.0f - lx)          * sig; if (!(vy1 & vx0)) w10 = 0.0f;
        float w11 = ly * lx                   * sig; if (!(vy1 & vx1)) w11 = 0.0f;
        const int yc0 = min(max(y0, 0), IMG_H - 1);
        const int yc1 = min(max(y1, 0), IMG_H - 1);
        const int xc0 = min(max(x0, 0), IMG_W - 1);
        const int xc1 = min(max(x1, 0), IMG_W - 1);
        const int base = (t * 32 + n) * 4;
        sPi[base + 0] = yc0 * IMG_W + xc0;
        sPi[base + 1] = yc0 * IMG_W + xc1;
        sPi[base + 2] = yc1 * IMG_W + xc0;
        sPi[base + 3] = yc1 * IMG_W + xc1;
        sPw[base + 0] = w00; sPw[base + 1] = w01;
        sPw[base + 2] = w10; sPw[base + 3] = w11;
    }
    __syncthreads();

    // ---- Phase B: gather 2 channels x 2 subtiles per thread, bf16 -> LDS ----
    {
        const int n  = tid & 15;
        const int c0 = (tid >> 4) << 1;                       // 0..62, even
        const float* xp0 = x + ((size_t)(b * CIN + c0) << 16);
        const float* xp1 = xp0 + NPIX;
        const int kh = c0 >> 5;
        const int hi = (c0 >> 3) & 1;
        const int e  = (c0 & 7) | ((c0 >> 1) & 8);            // even element
#pragma unroll
        for (int t = 0; t < 9; ++t) {
#pragma unroll
            for (int u = 0; u < 2; ++u) {
                const int pb = (t * 32 + u * 16 + n) * 4;
                const int o00 = sPi[pb + 0], o01 = sPi[pb + 1];
                const int o10 = sPi[pb + 2], o11 = sPi[pb + 3];
                const float w00 = sPw[pb + 0], w01 = sPw[pb + 1];
                const float w10 = sPw[pb + 2], w11 = sPw[pb + 3];
                float v0 = w00 * xp0[o00] + w01 * xp0[o01] + w10 * xp0[o10] + w11 * xp0[o11];
                float v1 = w00 * xp1[o00] + w01 * xp1[o01] + w10 * xp1[o10] + w11 * xp1[o11];
                const int lidx = ((((t * 2 + u) * 2 + kh) * 32) + hi * 16 + n) * 16 + e;
                sB[lidx]     = (__bf16)v0;
                sB[lidx + 1] = (__bf16)v1;
            }
        }
    }
    __syncthreads();

    // ---- Phase C: WMMA accumulate + rotation reduction ----
    {
        const int wave = tid >> 5;          // = M-tile index, 0..15
        const int lane = tid & 31;
        v8f acc0 = {0.f, 0.f, 0.f, 0.f, 0.f, 0.f, 0.f, 0.f};
        v8f acc1 = {0.f, 0.f, 0.f, 0.f, 0.f, 0.f, 0.f, 0.f};
#pragma unroll
        for (int t = 0; t < 9; ++t) {
#pragma unroll
            for (int h = 0; h < 2; ++h) {
                // One A-fragment load serves both N-subtiles (2 WMMAs).
                const v16bf a = *(const v16bf*)(Aw + (((((wave * 9) + t) * 2 + h) * 32 + lane) << 4));
                const v16bf b0 = *(const v16bf*)(&sB[((((t * 2 + 0) * 2 + h) * 32) + lane) << 4]);
                acc0 = __builtin_amdgcn_wmma_f32_16x16x32_bf16(
                    false, a, false, b0, (short)0, acc0, false, false);
                const v16bf b1 = *(const v16bf*)(&sB[((((t * 2 + 1) * 2 + h) * 32) + lane) << 4]);
                acc1 = __builtin_amdgcn_wmma_f32_16x16x32_bf16(
                    false, a, false, b1, (short)0, acc1, false, false);
            }
        }
        // C/D layout: VGPR j -> rows j (lanes 0-15) and j+8 (lanes 16-31);
        // rotation r = j&3 in both halves, o_local = j>>2 (+2 for hi lanes).
        const int o = (wave << 2) + ((lane >> 4) << 1);       // +0 for oa, +1 for ob
        const size_t outA = ((size_t)(b * COUT + o)     << 16);
        const size_t outB = ((size_t)(b * COUT + o + 1) << 16);
#pragma unroll
        for (int u = 0; u < 2; ++u) {
            const v8f acc = u ? acc1 : acc0;
            const int ppix = pix + u * 16 + (lane & 15);
            const float g0 = gm[0 * NPIX + ppix];
            const float g1 = gm[1 * NPIX + ppix];
            const float g2 = gm[2 * NPIX + ppix];
            const float g3 = gm[3 * NPIX + ppix];
            float oa = fmaxf(acc[0], 0.f) * g0 + fmaxf(acc[1], 0.f) * g1 +
                       fmaxf(acc[2], 0.f) * g2 + fmaxf(acc[3], 0.f) * g3;
            float ob = fmaxf(acc[4], 0.f) * g0 + fmaxf(acc[5], 0.f) * g1 +
                       fmaxf(acc[6], 0.f) * g2 + fmaxf(acc[7], 0.f) * g3;
            out[outA + ppix] = oa;
            out[outB + ppix] = ob;
        }
    }
}

// ---------------------------------------------------------------------------
// Workspace layout (bytes):
//   [0,        1048576)  : pooled gm, float[4*65536]
//   [1048576,  10485760) : offsets,   float[2*18*65536]
//   [10485760, 15204352) : sigmoid mask, float[2*9*65536]
//   [15204352, 15499264) : A fragments, bf16[16*9*2*32*16]
// ---------------------------------------------------------------------------
extern "C" void kernel_launch(void* const* d_in, const int* in_sizes, int n_in,
                              void* d_out, int out_size, void* d_ws, size_t ws_size,
                              hipStream_t stream) {
    const float* x      = (const float*)d_in[0];
    const float* weight = (const float*)d_in[1];
    const float* off_w  = (const float*)d_in[2];
    const float* off_b  = (const float*)d_in[3];
    const float* msk_w  = (const float*)d_in[4];
    const float* msk_b  = (const float*)d_in[5];
    const float* gmask  = (const float*)d_in[6];
    float* out = (float*)d_out;

    char* ws = (char*)d_ws;
    float*  gmw  = (float*)(ws + 0);
    float*  offs = (float*)(ws + 1048576);
    float*  dmw  = (float*)(ws + 10485760);
    __bf16* Aw   = (__bf16*)(ws + 15204352);

    pool_gm_kernel<<<(4 * NPIX) / 256, 256, 0, stream>>>(gmask, gmw);
    build_tw_kernel<<<(16 * 9 * 2 * 32 * 16) / 256, 256, 0, stream>>>(weight, Aw);
    conv_off_mask_kernel<<<(2 * NPIX) / 256, 256, 0, stream>>>(
        x, off_w, off_b, msk_w, msk_b, offs, dmw);
    rotdeform_main_kernel<<<(2 * NPIX) / 32, 512, 0, stream>>>(
        x, offs, dmw, Aw, gmw, out);
}
